// BinarizeLinearWithFoldedBN_70162585747567
// MI455X (gfx1250) — 8-bit matrix operand order repaired and checked by reading, compiled, not run on hardware
//
#include <hip/hip_runtime.h>
#include <stdint.h>

typedef __attribute__((ext_vector_type(16))) int   v16i;
typedef __attribute__((ext_vector_type(4)))  int   v4i;
typedef __attribute__((ext_vector_type(8)))  float v8f;

typedef __attribute__((address_space(1))) v4i as1_v4i;
typedef __attribute__((address_space(3))) v4i as3_v4i;

#define FP8_P1 0x38u
#define FP8_M1 0xB8u
#define BN_EPS 1e-5f

__device__ __forceinline__ void async_copy_b128(const void* g, void* l) {
#if __has_builtin(__builtin_amdgcn_global_load_async_to_lds_b128)
    __builtin_amdgcn_global_load_async_to_lds_b128((as1_v4i*)g, (as3_v4i*)l, 0, 0);
#else
    uint32_t loff = (uint32_t)(uintptr_t)(as3_v4i*)l;
    asm volatile("global_load_async_to_lds_b128 %0, %1, off"
                 :: "v"(loff), "v"((uint64_t)(uintptr_t)g) : "memory");
#endif
}

__device__ __forceinline__ void wait_async_zero() {
#if __has_builtin(__builtin_amdgcn_s_wait_asynccnt)
    __builtin_amdgcn_s_wait_asynccnt(0);
#else
    asm volatile("s_wait_asynccnt 0x0" ::: "memory");
#endif
}

__global__ __launch_bounds__(256) void binarize_fp8_kernel(
    const float* __restrict__ in, uint8_t* __restrict__ out, int n4)
{
    int i = blockIdx.x * 256 + threadIdx.x;
    if (i >= n4) return;
    float4 f = reinterpret_cast<const float4*>(in)[i];
    uint32_t b0 = f.x > 0.f ? FP8_P1 : (f.x < 0.f ? FP8_M1 : 0u);
    uint32_t b1 = f.y > 0.f ? FP8_P1 : (f.y < 0.f ? FP8_M1 : 0u);
    uint32_t b2 = f.z > 0.f ? FP8_P1 : (f.z < 0.f ? FP8_M1 : 0u);
    uint32_t b3 = f.w > 0.f ? FP8_P1 : (f.w < 0.f ? FP8_M1 : 0u);
    reinterpret_cast<uint32_t*>(out)[i] = b0 | (b1 << 8) | (b2 << 16) | (b3 << 24);
}

__global__ __launch_bounds__(256) void bingemm_bn_kernel(
    const uint8_t* __restrict__ A,
    const uint8_t* __restrict__ Bm,
    const float* __restrict__ gamma, const float* __restrict__ beta,
    const float* __restrict__ mean,  const float* __restrict__ var,
    float* __restrict__ out, int M, int N, int K)
{
    __shared__ __align__(16) uint8_t As[2][128 * 128];
    __shared__ __align__(16) uint8_t Bs[2][128 * 128];

    const int tid  = threadIdx.x;
    const int lane = tid & 31;
    const int wave = tid >> 5;
    const int wm   = wave >> 1;
    const int wn   = wave & 1;
    const int half = lane >> 4;
    const int l16  = lane & 15;

    const int rowBase = blockIdx.y * 128;
    const int colBase = blockIdx.x * 128;

    auto stage = [&](int kt, int buf) {
        const int k0 = kt * 128;
        #pragma unroll
        for (int i = 0; i < 4; ++i) {
            int idx = tid + i * 256;
            int r = idx >> 3;
            int c = (idx & 7) << 4;
            async_copy_b128(A  + (size_t)(rowBase + r) * K + k0 + c,
                            &As[buf][r * 128 + c]);
            async_copy_b128(Bm + (size_t)(colBase + r) * K + k0 + c,
                            &Bs[buf][r * 128 + c]);
        }
    };

    v8f acc[2][4] = {};

    const int T = K / 128;
    stage(0, 0);

    for (int t = 0; t < T; ++t) {
        const int cur = t & 1;
        wait_async_zero();
        __syncthreads();
        if (t + 1 < T) stage(t + 1, cur ^ 1);

        union { v16i v; unsigned long long d[8]; } af[2];
        #pragma unroll
        for (int tm = 0; tm < 2; ++tm) {
            const uint8_t* arow = &As[cur][(wm * 32 + tm * 16 + l16) * 128];
            #pragma unroll
            for (int blk = 0; blk < 2; ++blk) {
                #pragma unroll
                for (int g = 0; g < 4; ++g) {
                    af[tm].d[blk * 4 + g] = *reinterpret_cast<const unsigned long long*>(
                        arow + blk * 64 + g * 16 + half * 8);
                }
            }
        }

        #pragma unroll
        for (int tn = 0; tn < 4; ++tn) {
            union { v16i v; v4i q[4]; unsigned long long d[8]; } bf;
            const uint8_t* brow = &Bs[cur][(wn * 64 + tn * 16 + l16) * 128];
            #pragma unroll
            for (int g = 0; g < 8; ++g)
                bf.d[g] = *reinterpret_cast<const unsigned long long*>(brow + (g >> 2) * 64 + (g & 3) * 16 + half * 8);
            #pragma unroll
            for (int tm = 0; tm < 2; ++tm) {
                acc[tm][tn] = __builtin_amdgcn_wmma_f32_16x16x128_fp8_fp8(
                    af[tm].v, bf.v, (short)0, acc[tm][tn], false, false);
            }
        }
    }

    #pragma unroll
    for (int tn = 0; tn < 4; ++tn) {
        int o = colBase + wn * 64 + tn * 16 + l16;
        float scale = gamma[o] * rsqrtf(var[o] + BN_EPS);
        float shift = beta[o] - mean[o] * scale;
        #pragma unroll
        for (int tm = 0; tm < 2; ++tm) {
            int row0 = rowBase + wm * 32 + tm * 16 + half * 8;
            #pragma unroll
            for (int v = 0; v < 8; ++v) {
                out[(size_t)(row0 + v) * N + o] = acc[tm][tn][v] * scale + shift;
            }
        }
    }
}

extern "C" void kernel_launch(void* const* d_in, const int* in_sizes, int n_in,
                              void* d_out, int out_size, void* d_ws, size_t ws_size,
                              hipStream_t stream)
{
    const float* x      = (const float*)d_in[0];
    const float* weight = (const float*)d_in[1];
    const float* gamma  = (const float*)d_in[2];
    const float* beta   = (const float*)d_in[3];
    const float* mean   = (const float*)d_in[4];
    const float* var    = (const float*)d_in[5];
    float* out = (float*)d_out;

    const int OUT = in_sizes[2];
    const int IN  = in_sizes[1] / OUT;
    const int B   = in_sizes[0] / IN;

    uint8_t* xb = (uint8_t*)d_ws;
    uint8_t* wb = xb + (size_t)B * IN;

    int n4x = (B * IN) / 4;
    int n4w = (OUT * IN) / 4;
    binarize_fp8_kernel<<<(n4x + 255) / 256, 256, 0, stream>>>(x, xb, n4x);
    binarize_fp8_kernel<<<(n4w + 255) / 256, 256, 0, stream>>>(weight, wb, n4w);

    dim3 grid(OUT / 128, B / 128);
    bingemm_bn_kernel<<<grid, 256, 0, stream>>>(xb, wb, gamma, beta, mean, var,
                                                out, B, OUT, IN);
}
